// Indexer_87385404604996
// MI455X (gfx1250) — matrix operand order repaired and checked by reading, not compiled, not run on hardware
//
#include <hip/hip_runtime.h>
#include <hip/hip_bf16.h>
#include <math.h>

// ---------------- types ----------------
typedef __attribute__((ext_vector_type(16))) _Float16 v16h;
typedef __attribute__((ext_vector_type(8)))  _Float16 v8h;
typedef __attribute__((ext_vector_type(8)))  float    v8f;
typedef __attribute__((ext_vector_type(4)))  float    v4f;
typedef __attribute__((ext_vector_type(4)))  int      v4i;

// ---------------- problem constants ----------------
constexpr int T_TOK = 2048;
constexpr int NH    = 32;
constexpr int DH    = 128;
constexpr int HIDN  = 7168;
constexpr int QLRK  = 1536;
constexpr int QCOLS = NH * DH;   // 4096
constexpr int NTOPK = 512;
constexpr float SCALE_F = 0.015625f;           // D^-0.5 * H^-0.5 = 1/64
constexpr float NEG_INF = -3.402823466e38f;

// ---------------- async-to-LDS (verified present on this toolchain) ----------------
#if defined(__has_builtin)
# if __has_builtin(__builtin_amdgcn_global_load_async_to_lds_b128)
#  define HAVE_ASYNC_LDS 1
# endif
#endif
#ifndef HAVE_ASYNC_LDS
# define HAVE_ASYNC_LDS 0
#endif
#define ASG __attribute__((address_space(1)))
#define ASL __attribute__((address_space(3)))
typedef ASG v4i* gv4i_p;   // global-AS pointer to 16-byte vector
typedef ASL v4i* lv4i_p;   // LDS-AS pointer to 16-byte vector

__device__ __forceinline__ void wait_async0() {
#if HAVE_ASYNC_LDS
# if __has_builtin(__builtin_amdgcn_s_wait_asynccnt)
  __builtin_amdgcn_s_wait_asynccnt(0);
# else
  asm volatile("s_wait_asynccnt 0" ::: "memory");
# endif
#endif
}

// ---------------- WMMA helpers ----------------
__device__ __forceinline__ v8f wmma_f16(v16h a, v16h b, v8f c) {
  // (neg_a, A, neg_b, B, c_mod, C, reuse_a, reuse_b)
  return __builtin_amdgcn_wmma_f32_16x16x32_f16(false, a, false, b, (short)0, c,
                                                false, false);
}

// A fragment (16x32 f16), row-major source (rows = M, cols = K).
// Lanes 0-15: halves 0..7 = K k0..k0+7, halves 8..15 = K k0+16..k0+23;
// lanes 16-31 shifted by +8 in K.
__device__ __forceinline__ v16h load_a_f16(const _Float16* base, int ld, int row0,
                                           int k0, int lane) {
  int m  = lane & 15;
  int kb = k0 + ((lane & 16) ? 8 : 0);
  const _Float16* p = base + (size_t)(row0 + m) * ld + kb;
  union { v16h v; v8h h[2]; } u;
  u.h[0] = *(const v8h*)(p);
  u.h[1] = *(const v8h*)(p + 16);
  return u.v;
}

// B fragment (32x16 f16), source is "N-major": row n holds K contiguously
// (B[k][n] = src[n][k]). Lanes 0-15: K k0..k0+7, k0+16..k0+23; lanes 16-31 +8.
__device__ __forceinline__ v16h load_b_f16(const _Float16* base, int ld, int n0,
                                           int k0, int lane) {
  int n  = n0 + (lane & 15);
  int kb = k0 + ((lane & 16) ? 8 : 0);
  const _Float16* p = base + (size_t)n * ld + kb;
  union { v16h v; v8h h[2]; } u;
  u.h[0] = *(const v8h*)(p);
  u.h[1] = *(const v8h*)(p + 16);
  return u.v;
}

// ---------------- kernel 0: bulk f32 -> f16 conversion ----------------
// 8 contiguous elements per thread; n is always a multiple of 8.
__global__ void __launch_bounds__(256) cvt_f16_kernel(
    const float* __restrict__ src, _Float16* __restrict__ dst, long n) {
  long i = ((long)blockIdx.x * blockDim.x + threadIdx.x) * 8;
  if (i + 8 <= n) {
    v4f x0 = *(const v4f*)(src + i);
    v4f x1 = *(const v4f*)(src + i + 4);
    v8h o;
#pragma unroll
    for (int j = 0; j < 4; ++j) {
      o[j]     = (_Float16)x0[j];
      o[j + 4] = (_Float16)x1[j];
    }
    *(v8h*)(dst + i) = o;
  }
}

// ---------------- kernel 1: q = q_lora @ wq_b^T, fused NeoX rope ----------------
// grid (T/16, QCOLS/64), block = 1 wave. Wave computes a 16(t) x 64(n) tile
// with 4 accumulators; A fragment reused 4x. Tiles with (n0 % 128 == 0) cover
// head dims 0..63, so rope pairs (d, d+32) live in accumulators j and j+2 of
// the same lane -> rotate in registers. Odd tiles (dims 64..127) pass through.
__global__ void __launch_bounds__(32) gemm_q_kernel(
    const _Float16* __restrict__ q_lora, const _Float16* __restrict__ wq_b,
    const int* __restrict__ positions, _Float16* __restrict__ qrope) {
  int t0 = blockIdx.x * 16;
  int n0 = blockIdx.y * 64;
  int lane = threadIdx.x & 31;
  v8f c[4] = {};
  for (int k0 = 0; k0 < QLRK; k0 += 32) {
    v16h a = load_a_f16(q_lora, QLRK, t0, k0, lane);
#pragma unroll
    for (int j = 0; j < 4; ++j) {
      v16h b = load_b_f16(wq_b, QLRK, n0 + j * 16, k0, lane);
      c[j] = wmma_f16(a, b, c[j]);
    }
  }
  int n = lane & 15, hi = (lane >> 4) & 1;
  if ((n0 & 127) == 0) {   // uniform per wave: this tile holds head dims 0..63
    float inv0 = powf(10000.0f, -(float)(n)      * (1.0f / 32.0f));
    float inv1 = powf(10000.0f, -(float)(16 + n) * (1.0f / 32.0f));
#pragma unroll
    for (int r = 0; r < 8; ++r) {
      float pos = (float)positions[t0 + r + 8 * hi];
      float a0 = pos * inv0, a1 = pos * inv1;
      float cs0 = cosf(a0), sn0 = sinf(a0);
      float cs1 = cosf(a1), sn1 = sinf(a1);
      float x1 = c[0][r], x2 = c[2][r];
      c[0][r] = x1 * cs0 - x2 * sn0;
      c[2][r] = x1 * sn0 + x2 * cs0;
      x1 = c[1][r]; x2 = c[3][r];
      c[1][r] = x1 * cs1 - x2 * sn1;
      c[3][r] = x1 * sn1 + x2 * cs1;
    }
  }
#pragma unroll
  for (int j = 0; j < 4; ++j)
#pragma unroll
    for (int r = 0; r < 8; ++r)
      qrope[(size_t)(t0 + r + 8 * hi) * QCOLS + n0 + j * 16 + n] = (_Float16)c[j][r];
}

// ---------------- kernel 2: k = hidden @ wk^T ; w = hidden @ wp^T * SCALE ------
// grid (T/16, 2), block = 1 wave. Wave computes 16(t) x 80(n): n-tiles
// nt*5 .. nt*5+4 of the 160-wide [k | w] output; tiles 0..7 -> k, 8..9 -> w.
__global__ void __launch_bounds__(32) gemm_kw_kernel(
    const _Float16* __restrict__ hidden, const _Float16* __restrict__ wk,
    const _Float16* __restrict__ wp, float* __restrict__ kraw,
    float* __restrict__ wbuf) {
  int t0 = blockIdx.x * 16;
  int nt = blockIdx.y;
  int lane = threadIdx.x & 31;
  const _Float16* wbase[5];
  int nbase[5];
#pragma unroll
  for (int j = 0; j < 5; ++j) {
    int tile = nt * 5 + j;
    wbase[j] = (tile < 8) ? wk : wp;
    nbase[j] = (tile < 8) ? tile * 16 : (tile - 8) * 16;
  }
  v8f c[5] = {};
  for (int k0 = 0; k0 < HIDN; k0 += 32) {
    v16h a = load_a_f16(hidden, HIDN, t0, k0, lane);
#pragma unroll
    for (int j = 0; j < 5; ++j) {
      v16h b = load_b_f16(wbase[j], HIDN, nbase[j], k0, lane);
      c[j] = wmma_f16(a, b, c[j]);
    }
  }
  int n = lane & 15, hi = (lane >> 4) & 1;
#pragma unroll
  for (int j = 0; j < 5; ++j) {
    int tile = nt * 5 + j;
    if (tile < 8) {
#pragma unroll
      for (int r = 0; r < 8; ++r)
        kraw[(size_t)(t0 + r + 8 * hi) * DH + nbase[j] + n] = c[j][r];
    } else {
#pragma unroll
      for (int r = 0; r < 8; ++r)
        wbuf[(size_t)(t0 + r + 8 * hi) * NH + nbase[j] + n] = c[j][r] * SCALE_F;
    }
  }
}

// ---------------- kernel 3: layernorm + rope on k -> f16 ----------------
// grid T, block 128 (one row per block)
__global__ void __launch_bounds__(128) ln_rope_k_kernel(
    const float* __restrict__ kraw, const float* __restrict__ gamma,
    const float* __restrict__ beta, const int* __restrict__ positions,
    _Float16* __restrict__ krope) {
  __shared__ float sh[128], sh2[128], nrm[128];
  __shared__ float mu_s, rstd_s;
  int t = blockIdx.x;
  int d = threadIdx.x;
  float x = kraw[(size_t)t * DH + d];
  sh[d] = x;
  sh2[d] = x * x;
  __syncthreads();
  for (int st = 64; st > 0; st >>= 1) {
    if (d < st) { sh[d] += sh[d + st]; sh2[d] += sh2[d + st]; }
    __syncthreads();
  }
  if (d == 0) {
    float mu  = sh[0] * (1.0f / 128.0f);
    float var = sh2[0] * (1.0f / 128.0f) - mu * mu;
    mu_s = mu;
    rstd_s = rsqrtf(var + 1e-6f);
  }
  __syncthreads();
  float y = (x - mu_s) * rstd_s * gamma[d] + beta[d];
  nrm[d] = y;
  __syncthreads();
  float pos = (float)positions[t];
  float outv;
  if (d < 32) {
    float inv = powf(10000.0f, -(float)d * (1.0f / 32.0f));
    float ang = pos * inv;
    outv = nrm[d] * cosf(ang) - nrm[d + 32] * sinf(ang);
  } else if (d < 64) {
    int i = d - 32;
    float inv = powf(10000.0f, -(float)i * (1.0f / 32.0f));
    float ang = pos * inv;
    outv = nrm[i] * sinf(ang) + nrm[i + 32] * cosf(ang);
  } else {
    outv = y;
  }
  krope[(size_t)t * DH + d] = (_Float16)outv;
}

// ---------------- kernel 4: fused score ----------------
// scores[t,s] = sum_h relu(q[t,h,:].k[s,:]) * w[t,h], masked to [cu_ks, cu_ke)
// grid (T/16, T/128), block 256 (8 waves; wave w owns s-subtile w).
// k block + q head-group staged through async-to-LDS.
__global__ void __launch_bounds__(256) score_kernel(
    const _Float16* __restrict__ qrope, const _Float16* __restrict__ krope,
    const float* __restrict__ wbuf, const int* __restrict__ cu_ks,
    const int* __restrict__ cu_ke, float* __restrict__ scores) {
  __shared__ alignas(16) _Float16 k_lds[128 * DH];      // 32 KB: 128 s-rows
  __shared__ alignas(16) _Float16 q_lds[16 * 4 * DH];   // 16 KB: 4 heads x 16 t-rows
  __shared__ alignas(16) float    w_lds[16 * NH];       // 2 KB

  int t0 = blockIdx.x * 16;
  int sb = blockIdx.y * 128;
  int tid = threadIdx.x, lane = tid & 31, wave = tid >> 5;
  int hi = (lane >> 4) & 1;

  // stage 128x128 k block (contiguous 32 KB)
  {
    const _Float16* src = krope + (size_t)sb * DH;
#if HAVE_ASYNC_LDS
    for (int i = tid; i < 2048; i += 256)
      __builtin_amdgcn_global_load_async_to_lds_b128(
          (gv4i_p)(src + i * 8), (lv4i_p)(k_lds + i * 8), 0, 0);
#else
    for (int i = tid; i < 2048; i += 256)
      *(v8h*)(k_lds + i * 8) = *(const v8h*)(src + i * 8);
#endif
  }
  // stage 16x32 w tile (contiguous 2 KB)
  for (int i = tid; i < 16 * NH; i += 256) w_lds[i] = wbuf[(size_t)t0 * NH + i];
  wait_async0();
  __syncthreads();

  v8f acc = {};
  for (int hg = 0; hg < 8; ++hg) {          // 8 groups of 4 heads
    __syncthreads();                         // readers of q_lds done
    // stage q for heads hg*4..hg*4+3: 16 rows x 512 halves = 1024 16B chunks
#if HAVE_ASYNC_LDS
    for (int i = tid; i < 1024; i += 256) {
      int r = i >> 6, cc = i & 63;
      __builtin_amdgcn_global_load_async_to_lds_b128(
          (gv4i_p)(qrope + (size_t)(t0 + r) * QCOLS + hg * 512 + cc * 8),
          (lv4i_p)(q_lds + (size_t)r * 512 + cc * 8), 0, 0);
    }
    wait_async0();
#else
    for (int i = tid; i < 1024; i += 256) {
      int r = i >> 6, cc = i & 63;
      *(v8h*)(q_lds + (size_t)r * 512 + cc * 8) =
          *(const v8h*)(qrope + (size_t)(t0 + r) * QCOLS + hg * 512 + cc * 8);
    }
#endif
    __syncthreads();
#pragma unroll
    for (int hh = 0; hh < 4; ++hh) {
      v8f c = {};
#pragma unroll
      for (int ch = 0; ch < 4; ++ch) {
        v16h a = load_a_f16(q_lds, 4 * DH, 0, hh * DH + ch * 32, lane);
        v16h b = load_b_f16(k_lds, DH, wave * 16, ch * 32, lane);
        c = wmma_f16(a, b, c);
      }
      int h = hg * 4 + hh;
#pragma unroll
      for (int r = 0; r < 8; ++r) {
        float wv = w_lds[(r + 8 * hi) * NH + h];
        acc[r] += fmaxf(c[r], 0.0f) * wv;
      }
    }
  }

  int s = sb + wave * 16 + (lane & 15);
#pragma unroll
  for (int r = 0; r < 8; ++r) {
    int t = t0 + r + 8 * hi;
    bool ok = (s >= cu_ks[t]) && (s < cu_ke[t]);
    scores[(size_t)t * T_TOK + s] = ok ? acc[r] : NEG_INF;
  }
}

// ---------------- kernel 5: per-row top-512 via bitonic sort ----------------
// grid T, block 256; sorts 2048 (val,idx) pairs descending, idx-ascending ties
__global__ void __launch_bounds__(256) topk_kernel(
    const float* __restrict__ scores, float* __restrict__ vals_out,
    int* __restrict__ idx_out) {
  __shared__ float sv[T_TOK];
  __shared__ int   si[T_TOK];
  int t = blockIdx.x;
  for (int i = threadIdx.x; i < T_TOK; i += 256) {
    sv[i] = scores[(size_t)t * T_TOK + i];
    si[i] = i;
  }
  __syncthreads();
  for (int k = 2; k <= T_TOK; k <<= 1) {
    for (int j = k >> 1; j > 0; j >>= 1) {
      for (int a = threadIdx.x; a < T_TOK; a += 256) {
        int b = a ^ j;
        if (b > a) {
          bool up = ((a & k) == 0);  // "up" segments sort descending overall
          float va = sv[a], vb = sv[b];
          int ia = si[a], ib = si[b];
          bool a_first = (va > vb) || (va == vb && ia < ib);
          bool do_swap = up ? !a_first : a_first;
          if (do_swap) {
            sv[a] = vb; sv[b] = va;
            si[a] = ib; si[b] = ia;
          }
        }
      }
      __syncthreads();
    }
  }
  for (int i = threadIdx.x; i < NTOPK; i += 256) {
    vals_out[(size_t)t * NTOPK + i] = sv[i];
    idx_out[(size_t)t * NTOPK + i]  = si[i];
  }
}

// ---------------- launcher ----------------
extern "C" void kernel_launch(void* const* d_in, const int* in_sizes, int n_in,
                              void* d_out, int out_size, void* d_ws,
                              size_t ws_size, hipStream_t stream) {
  const float* hidden  = (const float*)d_in[0];
  const float* q_lora  = (const float*)d_in[1];
  const float* wq_b_w  = (const float*)d_in[2];
  const float* wk_w    = (const float*)d_in[3];
  const float* wp_w    = (const float*)d_in[4];
  const float* k_gamma = (const float*)d_in[5];
  const float* k_beta  = (const float*)d_in[6];
  const int* positions = (const int*)d_in[7];
  const int* cu_ks     = (const int*)d_in[8];
  const int* cu_ke     = (const int*)d_in[9];

  char* ws = (char*)d_ws;
  size_t off = 0;
  auto carve = [&](size_t bytes) -> void* {
    void* p = ws + off;
    off += (bytes + 255) & ~(size_t)255;
    return p;
  };
  const long n_ql  = (long)T_TOK * QLRK;
  const long n_wqb = (long)QCOLS * QLRK;
  const long n_hid = (long)T_TOK * HIDN;
  const long n_wk  = (long)DH * HIDN;
  const long n_wp  = (long)NH * HIDN;

  _Float16* ql16   = (_Float16*)carve((size_t)n_ql * 2);
  _Float16* wqb16  = (_Float16*)carve((size_t)n_wqb * 2);
  _Float16* hid16  = (_Float16*)carve((size_t)n_hid * 2);
  _Float16* wk16   = (_Float16*)carve((size_t)n_wk * 2);
  _Float16* wp16   = (_Float16*)carve((size_t)n_wp * 2);
  _Float16* qrope  = (_Float16*)carve((size_t)T_TOK * QCOLS * 2);
  float*    kraw   = (float*)carve((size_t)T_TOK * DH * 4);
  _Float16* krope  = (_Float16*)carve((size_t)T_TOK * DH * 2);
  float*    wbuf   = (float*)carve((size_t)T_TOK * NH * 4);
  float*    scores = (float*)carve((size_t)T_TOK * T_TOK * 4);

  float* vals = (float*)d_out;
  int*   idx  = (int*)d_out + (size_t)T_TOK * NTOPK;

  auto cvt = [&](const float* s, _Float16* d, long n) {
    cvt_f16_kernel<<<(unsigned)(n / 8 / 256), 256, 0, stream>>>(s, d, n);
  };
  cvt(q_lora, ql16, n_ql);
  cvt(wq_b_w, wqb16, n_wqb);
  cvt(hidden, hid16, n_hid);
  cvt(wk_w, wk16, n_wk);
  cvt(wp_w, wp16, n_wp);

  gemm_q_kernel<<<dim3(T_TOK / 16, QCOLS / 64), 32, 0, stream>>>(ql16, wqb16,
                                                                 positions, qrope);
  gemm_kw_kernel<<<dim3(T_TOK / 16, 2), 32, 0, stream>>>(hid16, wk16, wp16,
                                                         kraw, wbuf);
  ln_rope_k_kernel<<<T_TOK, 128, 0, stream>>>(kraw, k_gamma, k_beta, positions, krope);
  score_kernel<<<dim3(T_TOK / 16, T_TOK / 128), 256, 0, stream>>>(qrope, krope, wbuf,
                                                                  cu_ks, cu_ke, scores);
  topk_kernel<<<T_TOK, 256, 0, stream>>>(scores, vals, idx);
}
